// Model_730144440421
// MI455X (gfx1250) — matrix operand order repaired and checked by reading, not compiled, not run on hardware
//
#include <hip/hip_runtime.h>
#include <hip/hip_bf16.h>
#include <stdint.h>

// ---------------------------------------------------------------------------
// Problem dims
// ---------------------------------------------------------------------------
#define T_STEPS 1024
#define I_DIM   512
#define H_DIM   1024
#define G4      4096          // 4*H
#define KTOT    1536          // I + H  (concat [x_t | h])
#define BATCH   128
#define CDIM    1000
#define CPAD    1008          // 63 tiles of 16
#define NGRP    8             // batch groups (16 rows each)
#define NCOL    8             // col-group WGs per batch group

// Workspace layout (bytes), all 256B aligned
#define WC_OFF   0ull                 // bf16 [4096][1536]        = 12,582,912 B
#define BC_OFF   12582912ull          // f32  [4096]              =     16,384 B
#define WFC_OFF  12599296ull          // bf16 [1008][1024]        =  2,064,384 B
#define VB_OFF   14663680ull          // bf16 [8][2][16][1536]    =    786,432 B
#define CNT_OFF  15450112ull          // u32  [8] barrier counters

typedef __attribute__((ext_vector_type(16))) __bf16         v16bf;
typedef __attribute__((ext_vector_type(8)))  float          v8f;
typedef __attribute__((ext_vector_type(8)))  unsigned short u16x8;

union AF {
  v16bf          v;
  unsigned short us[16];
  u16x8          h2[2];
};

__device__ __forceinline__ unsigned short f2bf(float f) {
  union { float f; unsigned u; } v; v.f = f;
  unsigned r = v.u + 0x7FFFu + ((v.u >> 16) & 1u);   // round-to-nearest-even
  return (unsigned short)(r >> 16);
}
__device__ __forceinline__ float sigmoidf_(float x) { return 1.0f / (1.0f + __expf(-x)); }

// activation buffer: vbuf(group, parity) -> bf16 [16][KTOT]
__device__ __forceinline__ unsigned short* vbuf_ptr(unsigned char* ws, int g, int p) {
  return (unsigned short*)(ws + VB_OFF) + (size_t)((g * 2 + p) * 16) * KTOT;
}

// ---------------------------------------------------------------------------
// Init: pack Wc=[W_ih;W_hh] bf16, bc=b_ih+b_hh, Wfc bf16 (zero-padded),
// fill vbuf parity-0 with converted x_0 and zeroed h, zero barrier counters.
// ---------------------------------------------------------------------------
__global__ void init_kernel(const float* __restrict__ x,
                            const float* __restrict__ W_ih, const float* __restrict__ W_hh,
                            const float* __restrict__ b_ih, const float* __restrict__ b_hh,
                            const float* __restrict__ W_fc, unsigned char* __restrict__ ws) {
  unsigned short* Wc  = (unsigned short*)(ws + WC_OFF);
  float*          bc  = (float*)(ws + BC_OFF);
  unsigned short* Wfc = (unsigned short*)(ws + WFC_OFF);
  unsigned short* vb  = (unsigned short*)(ws + VB_OFF);
  unsigned int*   cnt = (unsigned int*)(ws + CNT_OFF);

  size_t tid    = (size_t)blockIdx.x * blockDim.x + threadIdx.x;
  size_t stride = (size_t)gridDim.x * blockDim.x;

  // combined recurrence weight, bf16, row-major [4096][1536]
  for (size_t i = tid; i < (size_t)G4 * KTOT; i += stride) {
    int g = (int)(i / KTOT), k = (int)(i % KTOT);
    float v = (k < I_DIM) ? W_ih[(size_t)g * I_DIM + k]
                          : W_hh[(size_t)g * H_DIM + (k - I_DIM)];
    Wc[i] = f2bf(v);
  }
  // combined bias
  for (size_t i = tid; i < (size_t)G4; i += stride) bc[i] = b_ih[i] + b_hh[i];
  // FC weight, bf16, padded to 1008 cols
  for (size_t i = tid; i < (size_t)CPAD * H_DIM; i += stride) {
    int c = (int)(i / H_DIM), k = (int)(i % H_DIM);
    Wfc[i] = (c < CDIM) ? f2bf(W_fc[(size_t)c * H_DIM + k]) : (unsigned short)0;
  }
  // vbuf parity 0: x part = x_0 (bf16), h part = 0
  for (size_t i = tid; i < (size_t)BATCH * I_DIM; i += stride) {
    int r = (int)(i >> 9), col = (int)(i & 511);
    int g = r >> 4, m = r & 15;
    vb[(size_t)((g * 2 + 0) * 16 + m) * KTOT + col] =
        f2bf(x[(size_t)r * T_STEPS * I_DIM + col]);
  }
  for (size_t i = tid; i < (size_t)NGRP * 16 * H_DIM; i += stride) {
    int g = (int)(i / (16 * H_DIM));
    int rem = (int)(i % (16 * H_DIM));
    int m = rem >> 10, j = rem & 1023;
    vb[(size_t)((g * 2 + 0) * 16 + m) * KTOT + I_DIM + j] = 0;
  }
  for (size_t i = tid; i < NGRP; i += stride) cnt[i] = 0;
}

// ---------------------------------------------------------------------------
// Persistent LSTM recurrence.
// 64 blocks: blockIdx = batch_group(0..7) * 8 + col_group(0..7).
// Wave w owns hidden slice j0 = col_group*128 + w*16 and its i/f/g/o tiles.
// Each WG also converts its 1/8 column slice of x_{t+1} into the next-parity
// activation buffer (pipelined, ordered by the per-step barrier) and
// prefetches x_{t+2}.
// ---------------------------------------------------------------------------
__global__ void __launch_bounds__(256, 1)
lstm_kernel(const float* __restrict__ x, unsigned char* __restrict__ ws) {
  const unsigned short* Wc = (const unsigned short*)(ws + WC_OFF);
  const float*          bc = (const float*)(ws + BC_OFF);
  unsigned int*         cnt = (unsigned int*)(ws + CNT_OFF);

  const int lane = threadIdx.x & 31;
  const int wave = threadIdx.x >> 5;
  const int bgrp = blockIdx.x >> 3;
  const int cgrp = blockIdx.x & 7;
  const int half = lane >> 4;
  const int l15  = lane & 15;

  const int j0   = cgrp * 128 + wave * 16;   // hidden slice base
  const int jcol = j0 + l15;                 // this lane's hidden column

  // B fragment base pointers for the 4 gates (16 contiguous K per lane)
  const unsigned short* bp0 = Wc + ((size_t)(0 * H_DIM + jcol)) * KTOT + half * 16;
  const unsigned short* bp1 = Wc + ((size_t)(1 * H_DIM + jcol)) * KTOT + half * 16;
  const unsigned short* bp2 = Wc + ((size_t)(2 * H_DIM + jcol)) * KTOT + half * 16;
  const unsigned short* bp3 = Wc + ((size_t)(3 * H_DIM + jcol)) * KTOT + half * 16;

  const float bi = bc[0 * H_DIM + jcol];
  const float bf = bc[1 * H_DIM + jcol];
  const float bg = bc[2 * H_DIM + jcol];
  const float bo = bc[3 * H_DIM + jcol];

  unsigned int* mycnt = cnt + bgrp;
  const int grpRow = bgrp * 16;

  v8f cst = {};   // cell state tile (f32, persistent in registers)

  for (int t = 0; t < T_STEPS; ++t) {
    const unsigned short* vbR = vbuf_ptr(ws, bgrp, t & 1);
    unsigned short*       vbW = vbuf_ptr(ws, bgrp, (t + 1) & 1);
    const unsigned short* ard = vbR + (size_t)l15 * KTOT + half * 16;

    v8f ai = {}, afv = {}, ag = {}, ao = {};

    // ---- uniform K loop: 48 chunks of 32 over [x_t | h] (all bf16) ----
    #pragma unroll 4
    for (int kk = 0; kk < 48; ++kk) {
      AF a;
      const unsigned short* ap = ard + kk * 32;
      a.h2[0] = *(const u16x8*)(ap);
      a.h2[1] = *(const u16x8*)(ap + 8);
      const int kb = kk * 32;
      v16bf b0 = *(const v16bf*)(bp0 + kb);
      v16bf b1 = *(const v16bf*)(bp1 + kb);
      v16bf b2 = *(const v16bf*)(bp2 + kb);
      v16bf b3 = *(const v16bf*)(bp3 + kb);
      ai  = __builtin_amdgcn_wmma_f32_16x16x32_bf16(false, a.v, false, b0, (short)0, ai,  false, false);
      afv = __builtin_amdgcn_wmma_f32_16x16x32_bf16(false, a.v, false, b1, (short)0, afv, false, false);
      ag  = __builtin_amdgcn_wmma_f32_16x16x32_bf16(false, a.v, false, b2, (short)0, ag,  false, false);
      ao  = __builtin_amdgcn_wmma_f32_16x16x32_bf16(false, a.v, false, b3, (short)0, ao,  false, false);
    }

    // ---- gate nonlinearities, cell/hidden update (wave-local) ----
    #pragma unroll
    for (int e = 0; e < 8; ++e) {
      float iv = sigmoidf_(ai[e]  + bi);
      float fv = sigmoidf_(afv[e] + bf);
      float gv = tanhf   (ag[e]  + bg);
      float ov = sigmoidf_(ao[e] + bo);
      float cv = fv * cst[e] + iv * gv;
      cst[e] = cv;
      float hv = ov * tanhf(cv);
      int m = e + half * 8;
      vbW[(size_t)m * KTOT + I_DIM + jcol] = f2bf(hv);
    }

    // ---- convert this WG's 1/8 col-slice of x_{t+1} into next buffer ----
    if (t + 1 < T_STEPS) {
      #pragma unroll
      for (int i = threadIdx.x; i < 1024; i += 256) {
        int row = i >> 6;
        int col = cgrp * 64 + (i & 63);
        float v = x[((size_t)(grpRow + row) * T_STEPS + (t + 1)) * I_DIM + col];
        vbW[(size_t)row * KTOT + col] = f2bf(v);
      }
    }
    // ---- prefetch this WG's slice of x_{t+2} (global_prefetch_b8) ----
    if (t + 2 < T_STEPS && threadIdx.x < 32) {
      int row   = threadIdx.x >> 1;
      int chunk = threadIdx.x & 1;
      const float* pf = x + ((size_t)(grpRow + row) * T_STEPS + (t + 2)) * I_DIM
                          + cgrp * 64 + chunk * 32;
      __builtin_prefetch(pf, 0, 3);
    }

    // ---- barrier among the 8 col-group WGs of this batch group ----
    __threadfence();          // make h / x-stage stores visible device-wide
    __syncthreads();
    if (threadIdx.x == 0) {
      __hip_atomic_fetch_add(mycnt, 1u, __ATOMIC_RELEASE, __HIP_MEMORY_SCOPE_AGENT);
      const unsigned int target = (unsigned)NCOL * (unsigned)(t + 1);
      while (__hip_atomic_load(mycnt, __ATOMIC_ACQUIRE, __HIP_MEMORY_SCOPE_AGENT) < target)
        __builtin_amdgcn_s_sleep(1);
    }
    __syncthreads();
    // per-thread acquire so every CU's L0 sees the fresh activations
    (void)__hip_atomic_load(mycnt, __ATOMIC_ACQUIRE, __HIP_MEMORY_SCOPE_AGENT);
  }
}

// ---------------------------------------------------------------------------
// Final FC: out[128][1000] = h_last @ W_fc^T + b_fc  (WMMA bf16, f32 acc).
// Final h is the h-part of vbuf parity 0 (t=1023 wrote parity (1023+1)&1==0).
// 63 blocks (col tiles) x 8 waves (row tiles).
// ---------------------------------------------------------------------------
__global__ void __launch_bounds__(256, 1)
fc_kernel(const float* __restrict__ b_fc, unsigned char* __restrict__ ws,
          float* __restrict__ out) {
  const unsigned short* Wfc = (const unsigned short*)(ws + WFC_OFF);

  const int lane = threadIdx.x & 31;
  const int wave = threadIdx.x >> 5;
  const int half = lane >> 4;
  const int l15  = lane & 15;
  const int m0   = wave * 16;
  const int n0   = blockIdx.x * 16;
  const int col  = n0 + l15;
  const int row  = m0 + l15;          // global batch row for A fragment

  const int g = row >> 4, m = row & 15;
  const unsigned short* ap = vbuf_ptr(ws, g, 0) + (size_t)m * KTOT + I_DIM + half * 16;
  const unsigned short* bp = Wfc + (size_t)col * H_DIM + half * 16;

  v8f acc = {};
  for (int kk = 0; kk < 32; ++kk) {
    AF a;
    a.h2[0] = *(const u16x8*)(ap + kk * 32);
    a.h2[1] = *(const u16x8*)(ap + kk * 32 + 8);
    v16bf b = *(const v16bf*)(bp + kk * 32);
    acc = __builtin_amdgcn_wmma_f32_16x16x32_bf16(false, a.v, false, b, (short)0, acc, false, false);
  }

  if (col < CDIM) {
    const float bias = b_fc[col];
    #pragma unroll
    for (int e = 0; e < 8; ++e) {
      const int mm = m0 + e + half * 8;
      out[(size_t)mm * CDIM + col] = acc[e] + bias;
    }
  }
}

// ---------------------------------------------------------------------------
extern "C" void kernel_launch(void* const* d_in, const int* in_sizes, int n_in,
                              void* d_out, int out_size, void* d_ws, size_t ws_size,
                              hipStream_t stream) {
  (void)in_sizes; (void)n_in; (void)out_size; (void)ws_size;
  const float* x    = (const float*)d_in[0];
  const float* W_ih = (const float*)d_in[1];
  const float* W_hh = (const float*)d_in[2];
  const float* b_ih = (const float*)d_in[3];
  const float* b_hh = (const float*)d_in[4];
  const float* W_fc = (const float*)d_in[5];
  const float* b_fc = (const float*)d_in[6];
  unsigned char* ws = (unsigned char*)d_ws;
  float*        out = (float*)d_out;

  hipLaunchKernelGGL(init_kernel, dim3(2048), dim3(256), 0, stream,
                     x, W_ih, W_hh, b_ih, b_hh, W_fc, ws);
  hipLaunchKernelGGL(lstm_kernel, dim3(64), dim3(256), 0, stream, x, ws);
  hipLaunchKernelGGL(fc_kernel, dim3(63), dim3(256), 0, stream, b_fc, ws, out);
}
